// StockHNet_2293512536850
// MI455X (gfx1250) — matrix operand order repaired and checked by reading, not compiled, not run on hardware
//
#include <hip/hip_runtime.h>
#include <hip/hip_bf16.h>

// ---------------------------------------------------------------------------
// Types for CDNA5 WMMA (wave32, 16x16x32 bf16, f32 accumulate) and TDM
// ---------------------------------------------------------------------------
typedef __attribute__((ext_vector_type(16))) __bf16 v16bf;
typedef __attribute__((ext_vector_type(8)))  float  v8f;
typedef __attribute__((ext_vector_type(4)))  unsigned int u32x4;
typedef __attribute__((ext_vector_type(8)))  int  i32x8;
typedef __attribute__((ext_vector_type(4)))  int  i32x4;

union Frag32B {           // 16 bf16 = 32 bytes = one WMMA A/B fragment
    v16bf v;
    uint4 q[2];
};

static constexpr int B_ = 8;
static constexpr int S_ = 4096;
static constexpr int D_ = 512;

__device__ __forceinline__ unsigned short f2bf(float f) {
    unsigned u = __builtin_bit_cast(unsigned, f);
    u += 0x7fffu + ((u >> 16) & 1u);          // round-to-nearest-even
    return (unsigned short)(u >> 16);
}

__device__ __forceinline__ float sigmoidf(float x) {
    return 1.0f / (1.0f + expf(-x));
}

__device__ __forceinline__ float wave_sum(float v) {
    for (int m = 16; m; m >>= 1) v += __shfl_xor(v, m, 32);
    return v;
}

// ---------------------------------------------------------------------------
// Kernel 0: transpose Wq/Wk (f32, [K][N]) -> bf16 [N][K]; zero atomic counter
// ---------------------------------------------------------------------------
__global__ __launch_bounds__(256) void prep_kernel(
    const float* __restrict__ Wq, const float* __restrict__ Wk,
    unsigned short* __restrict__ WqT, unsigned short* __restrict__ WkT,
    unsigned* __restrict__ counter) {
    int i = blockIdx.x * 256 + threadIdx.x;          // 0 .. 512*512-1
    int k = i >> 9;
    int n = i & (D_ - 1);
    WqT[n * D_ + k] = f2bf(Wq[i]);
    WkT[n * D_ + k] = f2bf(Wk[i]);
    if (i == 0) *counter = 0u;
}

// ---------------------------------------------------------------------------
// Kernel 1: Q/K GEMM, bf16 WMMA, LDS-resident operands.
//   grid = (M/64, N/64). block = 256 threads = 8 waves.
//   LDS: x tile 64x512 bf16 (64 KB) + Wq/Wk slices 64x512 bf16 (2x64 KB).
//   W slices staged via TDM tensor_load_to_lds (TENSORcnt); x is staged with
//   a f32->bf16 conversion pass. Inner loop: ds_load_b128 + v_wmma only.
//   Wave w: rows mtile = w>>1 (16 rows), cols (w&1)*32 .. +31 for both Q,K.
// ---------------------------------------------------------------------------
__global__ __launch_bounds__(256) void qk_gemm_kernel(
    const float* __restrict__ x,
    const unsigned short* __restrict__ WqT,
    const unsigned short* __restrict__ WkT,
    const float* __restrict__ bq, const float* __restrict__ bk,
    float* __restrict__ qout, float* __restrict__ kout) {

    __shared__ __align__(16) unsigned short lds_x [64 * D_];   // 64 KB
    __shared__ __align__(16) unsigned short lds_wq[64 * D_];   // 64 KB
    __shared__ __align__(16) unsigned short lds_wk[64 * D_];   // 64 KB

    const int tid  = threadIdx.x;
    const int lane = tid & 31;
    const int wv   = tid >> 5;
    const size_t m0 = (size_t)blockIdx.x * 64;
    const int    n0 = blockIdx.y * 64;

    // ---- Stage W slices (already bf16) into LDS ---------------------------
#if __has_builtin(__builtin_amdgcn_tensor_load_to_lds) && \
    __has_builtin(__builtin_amdgcn_s_wait_tensorcnt)
    if (wv == 0) {
        // Build D# (group0 + group1) for a 2D 64x512 bf16 tile copy.
        auto make_g0 = [](const unsigned short* g, unsigned lds_off) {
            unsigned long long ga = (unsigned long long)g;
            u32x4 g0;
            g0[0] = 1u;                                   // count=1, user D#
            g0[1] = lds_off;                              // lds_addr (bytes)
            g0[2] = (unsigned)(ga & 0xffffffffu);         // global_addr lo
            g0[3] = (unsigned)((ga >> 32) & 0x01ffffffu)  // global_addr hi
                  | 0x80000000u;                          // type = 2 (image)
            return g0;
        };
        i32x8 g1;
        g1[0] = 0x00010000;        // workgroup_mask=0, data_size=1 (2 bytes)
        g1[1] = (D_ << 16);        // tensor_dim0 = 512 (low 16 bits)
        g1[2] = (64 << 16);        // tensor_dim0 hi=0 | tensor_dim1 = 64
        g1[3] = (D_ << 16);        // tensor_dim1 hi=0 | tile_dim0 = 512
        g1[4] = 64;                // tile_dim1 = 64, tile_dim2 = 0
        g1[5] = D_;                // tensor_dim0_stride = 512 (lo 32)
        g1[6] = 0;
        g1[7] = 0;
        i32x4 z4 = {0, 0, 0, 0};
        i32x8 z8 = {0, 0, 0, 0, 0, 0, 0, 0};
        unsigned offq = (unsigned)(unsigned long long)
            (__attribute__((address_space(3))) unsigned short*)lds_wq;
        unsigned offk = (unsigned)(unsigned long long)
            (__attribute__((address_space(3))) unsigned short*)lds_wk;
        __builtin_amdgcn_tensor_load_to_lds(
            make_g0(WqT + (size_t)n0 * D_, offq), g1, z4, z4, z8, 0);
        __builtin_amdgcn_tensor_load_to_lds(
            make_g0(WkT + (size_t)n0 * D_, offk), g1, z4, z4, z8, 0);
        __builtin_amdgcn_s_wait_tensorcnt(0);
    }
#else
    {   // Fallback: cooperative 16-byte copies.
        const uint4* sq = (const uint4*)(WqT + (size_t)n0 * D_);
        const uint4* sk = (const uint4*)(WkT + (size_t)n0 * D_);
        uint4* dq = (uint4*)lds_wq;
        uint4* dk = (uint4*)lds_wk;
        #pragma unroll
        for (int i = 0; i < 16; ++i) {
            int c = tid + i * 256;               // 4096 chunks of 16 B
            dq[c] = sq[c];
            dk[c] = sk[c];
        }
    }
#endif

    // ---- Stage x tile (f32 -> bf16) into LDS ------------------------------
    {
        const float4* src = (const float4*)(x + m0 * D_);
        #pragma unroll
        for (int i = 0; i < 32; ++i) {
            int i4 = tid + i * 256;              // float4 index, 0..8191
            float4 f = src[i4];
            int e = i4 * 4;
            lds_x[e + 0] = f2bf(f.x);
            lds_x[e + 1] = f2bf(f.y);
            lds_x[e + 2] = f2bf(f.z);
            lds_x[e + 3] = f2bf(f.w);
        }
    }
    __syncthreads();

    const int hi    = lane >> 4;                 // 0: lanes 0-15, 1: 16-31
    const int nlo   = lane & 15;
    const int mtile = wv >> 1;                   // 0..3 (16 rows each)
    const int nhalf = wv & 1;                    // 0..1 (32 cols each)

    v8f accQ[2] = {};
    v8f accK[2] = {};

    #pragma unroll 4
    for (int ks = 0; ks < D_ / 32; ++ks) {
        // A fragment: row = mtile*16 + nlo; K contiguous at hi*16 .. hi*16+15.
        Frag32B a;
        const uint4* pa =
            (const uint4*)&lds_x[(mtile * 16 + nlo) * D_ + ks * 32 + hi * 16];
        a.q[0] = pa[0];
        a.q[1] = pa[1];

        #pragma unroll
        for (int nt = 0; nt < 2; ++nt) {
            const int nl = nhalf * 32 + nt * 16 + nlo;
            const uint4* pq =
                (const uint4*)&lds_wq[nl * D_ + ks * 32 + hi * 16];
            Frag32B fq; fq.q[0] = pq[0]; fq.q[1] = pq[1];
            const uint4* pk =
                (const uint4*)&lds_wk[nl * D_ + ks * 32 + hi * 16];
            Frag32B fk; fk.q[0] = pk[0]; fk.q[1] = pk[1];

            accQ[nt] = __builtin_amdgcn_wmma_f32_16x16x32_bf16(
                false, a.v, false, fq.v, (short)0, accQ[nt], false, false);
            accK[nt] = __builtin_amdgcn_wmma_f32_16x16x32_bf16(
                false, a.v, false, fk.v, (short)0, accK[nt], false, false);
        }
    }

    // C/D layout: VGPR r -> row r (lanes 0-15) or r+8 (lanes 16-31), col=nlo.
    #pragma unroll
    for (int nt = 0; nt < 2; ++nt) {
        const int n = n0 + nhalf * 32 + nt * 16 + nlo;
        const float bQ = bq[n];
        const float bK = bk[n];
        #pragma unroll
        for (int r = 0; r < 8; ++r) {
            const size_t m = m0 + (size_t)(mtile * 16 + hi * 8 + r);
            qout[m * D_ + n] = accQ[nt][r] + bQ;
            kout[m * D_ + n] = accK[nt][r] + bK;
        }
    }
}

// ---------------------------------------------------------------------------
// Kernel 2: per-position boundary probability. One wave per (b,s).
// ---------------------------------------------------------------------------
__global__ __launch_bounds__(256) void sim_kernel(
    const float* __restrict__ x,
    const float* __restrict__ q, const float* __restrict__ k,
    const float* __restrict__ w_vol, const float* __restrict__ b_vol,
    const float* __restrict__ w_volu, const float* __restrict__ b_volu,
    float* __restrict__ probs, unsigned* __restrict__ hard,
    unsigned* __restrict__ counter) {

    const int wave = threadIdx.x >> 5;
    const int lane = threadIdx.x & 31;
    const size_t idx = (size_t)blockIdx.x * 8 + wave;     // < B*S
    const int s = (int)(idx & (S_ - 1));

    float prob;
    if (s == 0) {
        prob = sigmoidf(1.0f);                             // combined[:,0] = 1
    } else {
        const float4* qp = (const float4*)(q + idx * D_);
        const float4* kp = (const float4*)(k + (idx - 1) * D_);
        const float4* xp = (const float4*)(x + idx * D_);
        const float4* wv = (const float4*)w_vol;
        const float4* wu = (const float4*)w_volu;
        float qk = 0.f, qq = 0.f, kk = 0.f, xv = 0.f, xu = 0.f;
        #pragma unroll
        for (int j = 0; j < 4; ++j) {
            const int e = lane + j * 32;                   // 128 float4 per row
            float4 qf = qp[e], kf = kp[e], xf = xp[e], wvf = wv[e], wuf = wu[e];
            qk += qf.x * kf.x + qf.y * kf.y + qf.z * kf.z + qf.w * kf.w;
            qq += qf.x * qf.x + qf.y * qf.y + qf.z * qf.z + qf.w * qf.w;
            kk += kf.x * kf.x + kf.y * kf.y + kf.z * kf.z + kf.w * kf.w;
            xv += xf.x * wvf.x + xf.y * wvf.y + xf.z * wvf.z + xf.w * wvf.w;
            xu += xf.x * wuf.x + xf.y * wuf.y + xf.z * wuf.z + xf.w * wuf.w;
        }
        qk = wave_sum(qk); qq = wave_sum(qq); kk = wave_sum(kk);
        xv = wave_sum(xv); xu = wave_sum(xu);

        const float qn = fmaxf(sqrtf(qq), 1e-8f);
        const float kn = fmaxf(sqrtf(kk), 1e-8f);
        const float sim = qk / (qn * kn);
        const float vol  = sigmoidf(xv + b_vol[0]);
        const float volu = sigmoidf(xu + b_volu[0]);
        prob = sigmoidf((1.0f - sim) + 0.3f * vol + 0.2f * volu);
    }

    if (lane == 0) {
        probs[idx] = prob;
        const unsigned h = prob > 0.5f ? 1u : 0u;
        hard[idx] = h;
        if (h) atomicAdd(counter, 1u);
    }
}

// ---------------------------------------------------------------------------
// Kernel 3: per-batch inclusive scan of hard flags -> positions (one wave/b).
// ---------------------------------------------------------------------------
__global__ __launch_bounds__(32) void scan_kernel(
    const unsigned* __restrict__ hard, int* __restrict__ pos) {
    const int b = blockIdx.x;
    const int lane = threadIdx.x;
    int running = 0;
    for (int c = 0; c < S_ / 32; ++c) {
        const int s = c * 32 + lane;
        int v = (int)hard[(size_t)b * S_ + s];
        #pragma unroll
        for (int d = 1; d < 32; d <<= 1) {
            int t = __shfl_up(v, d, 32);
            if (lane >= d) v += t;
        }
        pos[(size_t)b * S_ + s] = running + v - 1;         // cumsum - 1
        running += __shfl(v, 31, 32);
    }
}

// ---------------------------------------------------------------------------
// Kernel 4: zero the compressed output region (d_out is poisoned by harness).
// ---------------------------------------------------------------------------
__global__ __launch_bounds__(256) void zero_kernel(float4* __restrict__ p, size_t n4) {
    size_t i = (size_t)blockIdx.x * blockDim.x + threadIdx.x;
    const size_t stride = (size_t)gridDim.x * blockDim.x;
    const float4 z = make_float4(0.f, 0.f, 0.f, 0.f);
    for (; i < n4; i += stride) p[i] = z;
}

// ---------------------------------------------------------------------------
// Kernel 5: scatter selected x rows into compressed. One wave per (b,s).
// ---------------------------------------------------------------------------
__global__ __launch_bounds__(256) void scatter_kernel(
    const float* __restrict__ x, const unsigned* __restrict__ hard,
    const int* __restrict__ pos, float* __restrict__ comp) {
    const int wave = threadIdx.x >> 5;
    const int lane = threadIdx.x & 31;
    const size_t idx = (size_t)blockIdx.x * 8 + wave;
    if (!hard[idx]) return;
    const size_t b = idx >> 12;                            // / S_
    const size_t p = (size_t)pos[idx];
    const float4* src = (const float4*)(x + idx * D_);
    float4* dst = (float4*)(comp + (b * S_ + p) * D_);
    #pragma unroll
    for (int j = 0; j < 4; ++j) {
        const int e = lane + j * 32;
        dst[e] = src[e];
    }
}

// ---------------------------------------------------------------------------
// Kernel 6: ratio loss.
// ---------------------------------------------------------------------------
__global__ void finalize_kernel(const unsigned* __restrict__ counter,
                                const int* __restrict__ chunk_ratio,
                                float* __restrict__ loss) {
    const float actual = (float)(*counter) / (float)B_;
    const float target = (float)S_ / (float)(*chunk_ratio);
    const float d = actual - target;
    *loss = d * d;
}

// ---------------------------------------------------------------------------
// Launch
// ---------------------------------------------------------------------------
extern "C" void kernel_launch(void* const* d_in, const int* in_sizes, int n_in,
                              void* d_out, int out_size, void* d_ws, size_t ws_size,
                              hipStream_t stream) {
    (void)in_sizes; (void)n_in; (void)out_size; (void)ws_size;

    const float* x      = (const float*)d_in[0];
    const float* Wq     = (const float*)d_in[1];
    const float* bq     = (const float*)d_in[2];
    const float* Wk     = (const float*)d_in[3];
    const float* bk     = (const float*)d_in[4];
    const float* w_vol  = (const float*)d_in[5];
    const float* b_vol  = (const float*)d_in[6];
    const float* w_volu = (const float*)d_in[7];
    const float* b_volu = (const float*)d_in[8];
    const int*   ratio  = (const int*)d_in[9];

    const size_t M = (size_t)B_ * S_;                      // 32768

    // Workspace layout (~136 MB):
    char* ws = (char*)d_ws;
    unsigned short* WqT = (unsigned short*)ws;             // 512 KB
    unsigned short* WkT = WqT + (size_t)D_ * D_;           // 512 KB
    float* qbuf  = (float*)(ws + 2u * D_ * D_ * sizeof(unsigned short)); // 64 MB
    float* kbuf  = qbuf + M * D_;                          // 64 MB
    unsigned* hard = (unsigned*)(kbuf + M * D_);           // 128 KB
    int* pos       = (int*)(hard + M);                     // 128 KB
    unsigned* counter = (unsigned*)(pos + M);              // 4 B

    // Output layout: compressed [B*S*D] | boundary_probs [B*S] | ratio_loss [1]
    float* comp  = (float*)d_out;
    float* probs = comp + M * D_;
    float* loss  = probs + M;

    prep_kernel<<<(D_ * D_) / 256, 256, 0, stream>>>(Wq, Wk, WqT, WkT, counter);
    dim3 gemm_grid((unsigned)(M / 64), D_ / 64);
    qk_gemm_kernel<<<gemm_grid, 256, 0, stream>>>(x, WqT, WkT, bq, bk,
                                                  qbuf, kbuf);
    sim_kernel<<<(unsigned)(M / 8), 256, 0, stream>>>(x, qbuf, kbuf, w_vol, b_vol,
                                                      w_volu, b_volu, probs, hard,
                                                      counter);
    scan_kernel<<<B_, 32, 0, stream>>>(hard, pos);
    zero_kernel<<<4096, 256, 0, stream>>>((float4*)comp, M * D_ / 4);
    scatter_kernel<<<(unsigned)(M / 8), 256, 0, stream>>>(x, hard, pos, comp);
    finalize_kernel<<<1, 1, 0, stream>>>(counter, ratio, loss);
}
